// NXTentLoss_79139067396375
// MI455X (gfx1250) — matrix operand order repaired and checked by reading, not compiled, not run on hardware
//
#include <hip/hip_runtime.h>
#include <hip/hip_bf16.h>

typedef __attribute__((ext_vector_type(16))) _Float16 v16h;
typedef __attribute__((ext_vector_type(8)))  _Float16 v8h;
typedef __attribute__((ext_vector_type(8)))  float    v8f;

#define N2    8192   // total rows (2N)
#define KDIM  128
#define NPOS  4096   // positive index = row ^ 4096
#define COLS_PER_CHUNK 512
#define NT (COLS_PER_CHUNK / 16)             // 32 column tiles per chunk
#define NCHUNKS (N2 / COLS_PER_CHUNK)        // 16
#define ROWS_PER_BLOCK 128                   // 8 waves * 16 rows
#define NROWBLOCKS (N2 / ROWS_PER_BLOCK)     // 64
#define LDS_STRIDE 136                       // halves: 272B row stride, conflict-free b128
#define EXP2_SCALE 2.8853900817779268f       // 2/ln2 : exp(2x) = exp2(x*EXP2_SCALE)

struct __align__(8) h4 { _Float16 x, y, z, w; };

#if __has_builtin(__builtin_amdgcn_tensor_load_to_lds) && __has_builtin(__builtin_amdgcn_s_wait_tensorcnt)
#define HAVE_TDM 1
#else
#define HAVE_TDM 0
#endif

// Raw v_exp_f32: argument range here is [-2.89, 2.89], far from the subnormal
// guard band the libm exp2f code protects against, so skip the range fixup.
__device__ __forceinline__ float fast_exp2(float x) {
#if __has_builtin(__builtin_amdgcn_exp2f)
  return __builtin_amdgcn_exp2f(x);
#else
  return exp2f(x);
#endif
}

#if HAVE_TDM
typedef __attribute__((ext_vector_type(4))) unsigned int u32x4;
typedef __attribute__((ext_vector_type(8))) int          i32x8;
typedef __attribute__((ext_vector_type(4))) int          i32x4;

// Stage one 16-row x 128-half B panel (4KB) global->LDS via the Tensor Data
// Mover. D# pad: +16B after every 256B row -> 272B LDS stride (bank-friendly).
__device__ __forceinline__ void tdm_stage(const _Float16* gsrc, unsigned int lds_addr) {
  const unsigned long long ga = (unsigned long long)(size_t)gsrc;
  u32x4 g0 = {};
  g0[0] = 1u;                                    // count=1, user descriptor
  g0[1] = lds_addr;                              // LDS byte address
  g0[2] = (unsigned int)ga;                      // global_addr[31:0]
  g0[3] = (unsigned int)((ga >> 32) & 0x01ffffffull) | (2u << 30); // addr[56:32] | type=2
  i32x8 g1 = {};
  g1[0] = (int)((1u << 16)    // data_size = 2 bytes
              | (1u << 20)    // pad_enable
              | (5u << 22)    // pad_interval: 64 DWORDs (256B)
              | (3u << 25));  // pad_amount: 4 DWORDs (16B)
  g1[1] = (int)(128u << 16);  // tensor_dim0 = 128 (elements)
  g1[2] = (int)(16u  << 16);  // tensor_dim1 = 16 (tile rows)
  g1[3] = (int)(128u << 16);  // tile_dim0 = 128
  g1[4] = 16;                 // tile_dim1 = 16, tile_dim2 = 0
  g1[5] = 128;                // tensor_dim0_stride = 128 elements
  i32x4 gz = {};
#if __clang_major__ >= 23
  i32x8 gz8 = {};
  __builtin_amdgcn_tensor_load_to_lds(g0, g1, gz, gz, gz8, 0);
#else
  __builtin_amdgcn_tensor_load_to_lds(g0, g1, gz, gz, 0);
#endif
}
#endif

// ---------------------------------------------------------------------------
// Kernel 1: row-normalize x (f32 -> f16); zero rowsum accumulator and output.
// ---------------------------------------------------------------------------
__global__ __launch_bounds__(256) void nxtent_norm(const float* __restrict__ x,
                                                   _Float16* __restrict__ xn,
                                                   float* __restrict__ rowsum,
                                                   float* __restrict__ out) {
  const int tid  = blockIdx.x * 256 + threadIdx.x;
  const int lane = threadIdx.x & 31;
  const int row  = tid >> 5;

  const float4 v = ((const float4*)(x + row * KDIM))[lane];
  float s = v.x * v.x + v.y * v.y + v.z * v.z + v.w * v.w;
#pragma unroll
  for (int off = 16; off > 0; off >>= 1) s += __shfl_xor(s, off, 32);
  const float inv = rsqrtf(fmaxf(s, 1e-24f));

  h4 h;
  h.x = (_Float16)(v.x * inv);
  h.y = (_Float16)(v.y * inv);
  h.z = (_Float16)(v.z * inv);
  h.w = (_Float16)(v.w * inv);
  ((h4*)(xn + row * KDIM))[lane] = h;

  if (tid < N2) rowsum[tid] = 0.0f;
  if (tid == 0) out[0] = 0.0f;
}

// ---------------------------------------------------------------------------
// Kernel 2: fused Gram / exp / masked row-sum / positive capture.
// B panels staged into LDS (TDM double-buffered when available); A tiles in
// registers; 16x16 sim tiles via v_wmma_f32_16x16x32_f16.
// ---------------------------------------------------------------------------
__global__ __launch_bounds__(256) void nxtent_sim(const _Float16* __restrict__ xn,
                                                  float* __restrict__ rowsum,
                                                  float* __restrict__ posbuf) {
  __shared__ _Float16 sb[2][16][LDS_STRIDE];

  const int tid  = threadIdx.x;
  const int lane = tid & 31;
  const int wave = tid >> 5;
  const int mlo  = lane & 15;   // A: row in tile ; B/C: column in tile
  const int hi   = lane >> 4;   // lane half selector

  const int rowBase    = blockIdx.x * ROWS_PER_BLOCK + wave * 16;
  const int colStart   = blockIdx.y * COLS_PER_CHUNK;
  const int posRowBase = rowBase ^ NPOS;

  // --- A tile (rows rowBase..+15, K=128), held in registers ----------------
  // 16-bit A 16x32 layout: lane(hi): K = k0 + hi*8 + 0..7 , k0+16+hi*8 + 0..7
  v16h A[4];
  const _Float16* arow = xn + (size_t)(rowBase + mlo) * KDIM;
#pragma unroll
  for (int kk = 0; kk < 4; ++kk) {
    union { v16h v; v8h h[2]; } ua;
    ua.h[0] = *(const v8h*)(arow + kk * 32 + hi * 8);
    ua.h[1] = *(const v8h*)(arow + kk * 32 + 16 + hi * 8);
    A[kk] = ua.v;
  }

  float rsum[8], pv[8];
#pragma unroll
  for (int r = 0; r < 8; ++r) { rsum[r] = 0.0f; pv[r] = 0.0f; }

#if HAVE_TDM
  if (wave == 0)
    tdm_stage(xn + (size_t)colStart * KDIM, (unsigned int)(size_t)&sb[0][0][0]);
#endif

  for (int ct = 0; ct < NT; ++ct) {
    const int colBase = colStart + ct * 16;
#if HAVE_TDM
    const int cur = ct & 1;
    if (wave == 0) __builtin_amdgcn_s_wait_tensorcnt(0);
    __syncthreads();                          // publish buf[cur] to all waves
    if (wave == 0 && (ct + 1) < NT)           // overlap: stage next tile
      tdm_stage(xn + (size_t)(colBase + 16) * KDIM,
                (unsigned int)(size_t)&sb[cur ^ 1][0][0]);
#else
    const int cur = 0;
    __syncthreads();
    *(v8h*)&sb[0][tid >> 4][(tid & 15) * 8] =
        *(const v8h*)(xn + (size_t)(colBase + (tid >> 4)) * KDIM + (tid & 15) * 8);
    __syncthreads();
#endif

    // Load all 4 B chunks first (one ds clause, one wait), then run the 4
    // WMMAs back-to-back so the XDL pipe stays fed.
    v16h B[4];
#pragma unroll
    for (int kk = 0; kk < 4; ++kk) {
      const _Float16* bp = &sb[cur][mlo][kk * 32 + hi * 8];
      union { v16h v; v8h h[2]; } ub;
      ub.h[0] = *(const v8h*)(bp);
      ub.h[1] = *(const v8h*)(bp + 16);
      B[kk] = ub.v;
    }
    v8f c = {};
#pragma unroll
    for (int kk = 0; kk < 4; ++kk) {
      c = __builtin_amdgcn_wmma_f32_16x16x32_f16(
          false, A[kk], false, B[kk], (short)0, c, false, false);
    }

    // Hot path: exp(sim/T) and row-sum accumulate (T = 0.5 fused into exp2).
    float e[8];
#pragma unroll
    for (int r = 0; r < 8; ++r) {
      e[r] = fast_exp2(c[r] * EXP2_SCALE);
      rsum[r] += e[r];
    }
    // Wave-uniform corrections: each fires for at most one tile per chunk.
    if (colBase == rowBase) {                 // remove self-similarity term
      const int d = mlo - hi * 8;
#pragma unroll
      for (int r = 0; r < 8; ++r) rsum[r] -= (d == r) ? e[r] : 0.0f;
    }
    if (colBase == posRowBase) {              // capture sim at positive pair
      const int d = mlo - hi * 8;
#pragma unroll
      for (int r = 0; r < 8; ++r) pv[r] += (d == r) ? c[r] : 0.0f;
    }
  }

  // Reduce across the 16 lanes of each half (each half = one set of 8 rows).
#pragma unroll
  for (int r = 0; r < 8; ++r) {
    float a = rsum[r], p = pv[r];
#pragma unroll
    for (int off = 1; off < 16; off <<= 1) {
      a += __shfl_xor(a, off, 32);
      p += __shfl_xor(p, off, 32);
    }
    rsum[r] = a; pv[r] = p;
  }

  if (mlo == 0) {                 // lanes 0 (rows +0..7) and 16 (rows +8..15)
    const int rb = rowBase + hi * 8;
#pragma unroll
    for (int r = 0; r < 8; ++r) atomicAdd(&rowsum[rb + r], rsum[r]);
    if (posRowBase >= colStart && posRowBase < colStart + COLS_PER_CHUNK) {
#pragma unroll
      for (int r = 0; r < 8; ++r) posbuf[rb + r] = pv[r];
    }
  }
}

// ---------------------------------------------------------------------------
// Kernel 3: loss = sum_i ( log(rowsum[i]) - 2 * sim_pos[i] )
// ---------------------------------------------------------------------------
__global__ __launch_bounds__(256) void nxtent_loss(const float* __restrict__ rowsum,
                                                   const float* __restrict__ posbuf,
                                                   float* __restrict__ out) {
  __shared__ float sh[256];
  float acc = 0.0f;
  for (int i = threadIdx.x; i < N2; i += 256)
    acc += logf(rowsum[i]) - 2.0f * posbuf[i];
  sh[threadIdx.x] = acc;
  __syncthreads();
  for (int s = 128; s > 0; s >>= 1) {
    if (threadIdx.x < s) sh[threadIdx.x] += sh[threadIdx.x + s];
    __syncthreads();
  }
  if (threadIdx.x == 0) out[0] = sh[0];
}

extern "C" void kernel_launch(void* const* d_in, const int* in_sizes, int n_in,
                              void* d_out, int out_size, void* d_ws, size_t ws_size,
                              hipStream_t stream) {
  (void)in_sizes; (void)n_in; (void)out_size; (void)ws_size;
  const float* x = (const float*)d_in[0];
  float* out = (float*)d_out;

  // workspace layout: [0,8192) rowsum f32 | [8192,16384) pos f32 | @64KB xn f16
  float* rowsum = (float*)d_ws;
  float* posbuf = rowsum + N2;
  _Float16* xn  = (_Float16*)((char*)d_ws + 65536);

  nxtent_norm<<<N2 / 8, 256, 0, stream>>>(x, xn, rowsum, out);
  nxtent_sim<<<dim3(NROWBLOCKS, NCHUNKS), 256, 0, stream>>>(xn, rowsum, posbuf);
  nxtent_loss<<<1, 256, 0, stream>>>(rowsum, posbuf, out);
}
